// DGCNN_net_28810640621724
// MI455X (gfx1250) — matrix operand order repaired and checked by reading, not compiled, not run on hardware
//
#include <hip/hip_runtime.h>
#include <hip/hip_bf16.h>

#define FMAXV 3.402823466e38f
#define LRELU(x) ((x) > 0.f ? (x) : 0.2f * (x))

typedef _Float16 v16h __attribute__((ext_vector_type(16)));
typedef _Float16 h8   __attribute__((ext_vector_type(8)));
typedef float    v8f  __attribute__((ext_vector_type(8)));
typedef int      v4i  __attribute__((__vector_size__(4 * sizeof(int))));

#if defined(__has_builtin)
#if __has_builtin(__builtin_amdgcn_global_load_async_to_lds_b128)
#define HAVE_ASYNC_LDS 1
#endif
#endif
#ifndef HAVE_ASYNC_LDS
#define HAVE_ASYNC_LDS 0
#endif

#if HAVE_ASYNC_LDS
// async 16B global -> LDS copy (ASYNCcnt-tracked, bypasses VGPRs)
#define ASYNC_CP16(gsrc, ldst)                                              \
  __builtin_amdgcn_global_load_async_to_lds_b128(                           \
      (__attribute__((address_space(1))) v4i*)(gsrc),                       \
      (__attribute__((address_space(3))) v4i*)(ldst), 0, 0)
#endif

__device__ __forceinline__ void async_wait0() {
#if HAVE_ASYNC_LDS
#if defined(__has_builtin) && __has_builtin(__builtin_amdgcn_s_wait_asynccnt)
  __builtin_amdgcn_s_wait_asynccnt(0);
#else
  asm volatile("s_wait_asynccnt 0" ::: "memory");
#endif
#endif
}

__device__ __forceinline__ v16h cat8(h8 lo, h8 hi) {
  return __builtin_shufflevector(lo, hi, 0,1,2,3,4,5,6,7,8,9,10,11,12,13,14,15);
}

__device__ __forceinline__ void atomicMaxF(float* addr, float val) {
  unsigned int* ai = (unsigned int*)addr;
  unsigned int old = *ai;
  while (__uint_as_float(old) < val) {
    unsigned int assumed = old;
    old = atomicCAS(ai, assumed, __float_as_uint(val));
    if (old == assumed) break;
  }
}

// ---------------------------------------------------------------------------
// x [B,3,N] -> xt3 [B*N,3] (point-major) + squared norms
// ---------------------------------------------------------------------------
__global__ void prep_kernel(const float* __restrict__ x, float* __restrict__ xt3,
                            float* __restrict__ norms, int N) {
  int i = blockIdx.x * blockDim.x + threadIdx.x;       // over B*N
  if (i >= 4 * N) return;
  int b = i / N, n = i % N;
  float s = 0.f;
  #pragma unroll
  for (int c = 0; c < 3; c++) {
    float v = x[((size_t)b * 3 + c) * N + n];
    xt3[(size_t)i * 3 + c] = v;
    s += v * v;
  }
  norms[i] = s;
}

__global__ void norms_kernel(const float* __restrict__ xin, int stride, int C,
                             float* __restrict__ norms, int N) {
  int i = blockIdx.x * blockDim.x + threadIdx.x;       // over B*N
  if (i >= 4 * N) return;
  const float* p = xin + (size_t)i * stride;
  float s = 0.f;
  for (int c = 0; c < C; c++) s += p[c] * p[c];
  norms[i] = s;
}

__global__ void init_neg_kernel(float* __restrict__ p, int n) {
  int i = blockIdx.x * blockDim.x + threadIdx.x;
  if (i < n) p[i] = -FMAXV;
}

// f32 weights [O][Cin] -> padded f16 [O][PC] (PC >= Cin, zero fill)
__global__ void wcvt_kernel(const float* __restrict__ src, _Float16* __restrict__ dst,
                            int O, int Cin, int PC) {
  int i = blockIdx.x * blockDim.x + threadIdx.x;
  if (i >= O * PC) return;
  int o = i / PC, c = i % PC;
  dst[i] = (c < Cin) ? (_Float16)src[(size_t)o * Cin + c] : (_Float16)0.f;
}

// ---------------------------------------------------------------------------
// KNN: one thread per query point; top-32 smallest squared distances.
// Query row cached in LDS [C][64] (bank-conflict-free); candidate rows are
// block-uniform global loads (L2-resident: whole feature map << 192MB L2).
// ---------------------------------------------------------------------------
__global__ void __launch_bounds__(64)
knn_kernel(const float* __restrict__ xin, int stride, int C,
           const float* __restrict__ norms, int* __restrict__ idx, int N) {
  __shared__ float qls[128 * 64];   // [C][64], C <= 128
  __shared__ float bd[32 * 64];     // top-K dists, slot-major
  __shared__ int   bi[32 * 64];     // top-K indices

  const int b = blockIdx.y;
  const int t = threadIdx.x;
  const int q = blockIdx.x * 64 + t;
  const float* xb = xin + (size_t)b * N * stride;
  const float* nb = norms + (size_t)b * N;

  for (int c = 0; c < C; c++) qls[c * 64 + t] = xb[(size_t)q * stride + c];
  #pragma unroll
  for (int s = 0; s < 32; s++) bd[s * 64 + t] = FMAXV;

  const float nq = nb[q];
  float worst = FMAXV;
  int   wslot = 0;

  for (int m = 0; m < N; m++) {
    const float* xm = xb + (size_t)m * stride;
    float dot = 0.f;
    for (int c = 0; c < C; c++) dot += qls[c * 64 + t] * xm[c];
    float d = nq + nb[m] - 2.f * dot;
    if (d < worst) {
      bd[wslot * 64 + t] = d;
      bi[wslot * 64 + t] = m;
      worst = -FMAXV;
      #pragma unroll
      for (int s = 0; s < 32; s++) {
        float v = bd[s * 64 + t];
        if (v > worst) { worst = v; wslot = s; }
      }
    }
  }
  #pragma unroll
  for (int s = 0; s < 32; s++)
    idx[((size_t)b * N + q) * 32 + s] = bi[s * 64 + t];
}

// ---------------------------------------------------------------------------
// EdgeConv fused: gather(nbr-ctr,ctr) -> f16 LDS -> WMMA GEMM with W ->
// BN + LeakyReLU -> max over K=32 neighbors.
// Block: 128 thr = 4 waves. 64 rows (2 points x 32 nbrs) x 64 out channels.
// Wave w owns row tile [16w,16w+16), 4 col tiles, acc in f32.
// Weights pre-converted to padded f16 [O][PC]; B tiles staged to LDS with
// GLOBAL_LOAD_ASYNC_TO_LDS_B128 (ASYNCcnt) when available.
// ---------------------------------------------------------------------------
template <int TWO_C>
__global__ void __launch_bounds__(128)
edgeconv_wmma(const float* __restrict__ xin, int in_stride,
              const int* __restrict__ idx,
              const _Float16* __restrict__ Wh, const float* __restrict__ g,
              const float* __restrict__ bn_b,
              float* __restrict__ out, int out_stride, int N) {
  constexpr int C   = TWO_C / 2;
  constexpr int CK  = 64;                    // k-chunk (channels) staged in LDS
  constexpr int AST = CK + 8;                // LDS row stride in halfs (16B aligned, skewed)
  constexpr int PC  = (TWO_C + 63) & ~63;    // padded weight stride

  __shared__ alignas(16) _Float16 Alds[64 * AST];
  __shared__ alignas(16) _Float16 Blds[64 * AST];
  __shared__ float scale_s[64], shift_s[64];
  __shared__ float pmax[4][64];
  __shared__ int   nbr[64];

  const int b    = blockIdx.z;
  const int p0   = blockIdx.x * 2;
  const int o0   = blockIdx.y * 64;
  const int tid  = threadIdx.x;
  const int wave = tid >> 5;
  const int lane = tid & 31;

  if (tid < 64) {
    scale_s[tid] = g[o0 + tid] * rsqrtf(1.0f + 1e-5f);
    shift_s[tid] = bn_b[o0 + tid];
    int p = p0 + (tid >> 5);
    nbr[tid] = idx[((size_t)b * N + p) * 32 + (tid & 31)];
  }
  __syncthreads();

  const float* xb = xin + (size_t)b * N * in_stride;

  v8f acc[4];
  #pragma unroll
  for (int i = 0; i < 4; i++)
    #pragma unroll
    for (int j = 0; j < 8; j++) acc[i][j] = 0.f;

  for (int c0 = 0; c0 < TWO_C; c0 += CK) {
    // B: weight tile, f16, async DMA to LDS (row o, col c -> B elem (k=c, n=o))
#if HAVE_ASYNC_LDS
    for (int t = tid; t < 64 * (CK / 8); t += 128) {     // 16B chunks
      int o = t >> 3, c16 = (t & 7) << 3;
      ASYNC_CP16(Wh + (size_t)(o0 + o) * PC + c0 + c16, &Blds[o * AST + c16]);
    }
#else
    for (int t = tid; t < 64 * CK; t += 128) {
      int o = t / CK, c = t % CK;
      Blds[o * AST + c] = Wh[(size_t)(o0 + o) * PC + c0 + c];
    }
#endif
    // A: edge features, gathered + converted f32 -> f16
    for (int t = tid; t < 64 * CK; t += 128) {
      int r = t / CK, c = t % CK;
      int cg = c0 + c;
      int p = p0 + (r >> 5);
      float v = 0.f;
      if (cg < C)
        v = xb[(size_t)nbr[r] * in_stride + cg] - xb[(size_t)p * in_stride + cg];
      else if (cg < TWO_C)
        v = xb[(size_t)p * in_stride + (cg - C)];
      Alds[r * AST + c] = (_Float16)v;
    }
    async_wait0();
    __syncthreads();

    const int m = lane & 15, hh = lane >> 4;
    #pragma unroll
    for (int ks = 0; ks < 2; ks++) {   // two 32-wide k-steps per chunk
      const int kc = ks * 32;
      // A frag: lane (m,hh), elems 0..7 -> k=hh*8+e, 8..15 -> k=16+hh*8+e
      h8 a0 = *(const h8*)&Alds[(wave * 16 + m) * AST + kc + hh * 8];
      h8 a1 = *(const h8*)&Alds[(wave * 16 + m) * AST + kc + 16 + hh * 8];
      v16h af = cat8(a0, a1);
      #pragma unroll
      for (int ct = 0; ct < 4; ct++) {
        // B frag: lane (n=m,hh), elems 0..7 -> k=hh*8+e, 8..15 -> k=16+hh*8+e
        h8 b0 = *(const h8*)&Blds[(ct * 16 + m) * AST + kc + hh * 8];
        h8 b1 = *(const h8*)&Blds[(ct * 16 + m) * AST + kc + 16 + hh * 8];
        v16h bf = cat8(b0, b1);
        acc[ct] = __builtin_amdgcn_wmma_f32_16x16x32_f16(
            false, af, false, bf, (short)0, acc[ct], false, false);
      }
    }
    __syncthreads();
  }

  // BN + LeakyReLU + max over this wave's 16 rows (D: lane col = lane&15,
  // reg r holds row wave*16 + r + 8*(lane>>4))
  {
    const int m = lane & 15;
    #pragma unroll
    for (int ct = 0; ct < 4; ct++) {
      int oc = ct * 16 + m;
      float sc = scale_s[oc], sh = shift_s[oc];
      float mx = -FMAXV;
      #pragma unroll
      for (int r = 0; r < 8; r++) {
        float hv = acc[ct][r] * sc + sh;
        mx = fmaxf(mx, LRELU(hv));
      }
      mx = fmaxf(mx, __shfl_xor(mx, 16));   // combine row halves (wave32)
      if (lane < 16) pmax[wave][oc] = mx;
    }
  }
  __syncthreads();
  {
    int pt = tid >> 6, col = tid & 63;      // 2 points x 64 cols
    float mv = fmaxf(pmax[2 * pt][col], pmax[2 * pt + 1][col]);
    out[((size_t)b * N + p0 + pt) * out_stride + (o0 + col)] = mv;
  }
}

// ---------------------------------------------------------------------------
// W5 projection: cat[B*N,512] x W5^T[512,1024], BN + LeakyReLU, fused
// block-max over 64 rows + global atomic max -> gmax[B,1024]
// ---------------------------------------------------------------------------
__global__ void __launch_bounds__(128)
gemm5_wmma(const float* __restrict__ catf, const _Float16* __restrict__ Wh5,
           const float* __restrict__ g5, const float* __restrict__ b5,
           float* __restrict__ gmax, int N) {
  constexpr int KD = 512, CK = 64, AST = CK + 8;
  __shared__ alignas(16) _Float16 Alds[64 * AST];
  __shared__ alignas(16) _Float16 Blds[64 * AST];
  __shared__ float scale_s[64], shift_s[64];
  __shared__ float pmax[4][64];

  const int row0 = blockIdx.x * 64;   // over B*N; 64 | N so one batch per block
  const int o0   = blockIdx.y * 64;
  const int b    = row0 / N;
  const int tid  = threadIdx.x, wave = tid >> 5, lane = tid & 31;

  if (tid < 64) {
    scale_s[tid] = g5[o0 + tid] * rsqrtf(1.0f + 1e-5f);
    shift_s[tid] = b5[o0 + tid];
  }
  __syncthreads();

  v8f acc[4];
  #pragma unroll
  for (int i = 0; i < 4; i++)
    #pragma unroll
    for (int j = 0; j < 8; j++) acc[i][j] = 0.f;

  for (int c0 = 0; c0 < KD; c0 += CK) {
#if HAVE_ASYNC_LDS
    for (int t = tid; t < 64 * (CK / 8); t += 128) {     // 16B chunks
      int o = t >> 3, c16 = (t & 7) << 3;
      ASYNC_CP16(Wh5 + (size_t)(o0 + o) * KD + c0 + c16, &Blds[o * AST + c16]);
    }
#else
    for (int t = tid; t < 64 * CK; t += 128) {
      int o = t / CK, c = t % CK;
      Blds[o * AST + c] = Wh5[(size_t)(o0 + o) * KD + c0 + c];
    }
#endif
    for (int t = tid; t < 64 * CK; t += 128) {
      int r = t / CK, c = t % CK;
      Alds[r * AST + c] = (_Float16)catf[(size_t)(row0 + r) * KD + c0 + c];
    }
    async_wait0();
    __syncthreads();

    const int m = lane & 15, hh = lane >> 4;
    #pragma unroll
    for (int ks = 0; ks < 2; ks++) {
      const int kc = ks * 32;
      h8 a0 = *(const h8*)&Alds[(wave * 16 + m) * AST + kc + hh * 8];
      h8 a1 = *(const h8*)&Alds[(wave * 16 + m) * AST + kc + 16 + hh * 8];
      v16h af = cat8(a0, a1);
      #pragma unroll
      for (int ct = 0; ct < 4; ct++) {
        h8 b0 = *(const h8*)&Blds[(ct * 16 + m) * AST + kc + hh * 8];
        h8 b1 = *(const h8*)&Blds[(ct * 16 + m) * AST + kc + 16 + hh * 8];
        v16h bf = cat8(b0, b1);
        acc[ct] = __builtin_amdgcn_wmma_f32_16x16x32_f16(
            false, af, false, bf, (short)0, acc[ct], false, false);
      }
    }
    __syncthreads();
  }

  {
    const int m = lane & 15;
    #pragma unroll
    for (int ct = 0; ct < 4; ct++) {
      int oc = ct * 16 + m;
      float sc = scale_s[oc], sh = shift_s[oc];
      float mx = -FMAXV;
      #pragma unroll
      for (int r = 0; r < 8; r++) {
        float hv = acc[ct][r] * sc + sh;
        mx = fmaxf(mx, LRELU(hv));
      }
      mx = fmaxf(mx, __shfl_xor(mx, 16));
      if (lane < 16) pmax[wave][oc] = mx;
    }
  }
  __syncthreads();
  if (tid < 64) {
    float mv = fmaxf(fmaxf(pmax[0][tid], pmax[1][tid]),
                     fmaxf(pmax[2][tid], pmax[3][tid]));
    atomicMaxF(&gmax[(size_t)b * 1024 + o0 + tid], mv);
  }
}

// ---------------------------------------------------------------------------
// Head MLP (B=4 rows): [h|lf|nf](1152) -> 512 -> 256 -> 28. One block per b.
// ---------------------------------------------------------------------------
__global__ void __launch_bounds__(256)
head_kernel(const float* __restrict__ gmax,
            const float* __restrict__ l, const float* __restrict__ n,
            const float* __restrict__ W6, const float* __restrict__ g6, const float* __restrict__ b6,
            const float* __restrict__ W7, const float* __restrict__ g7, const float* __restrict__ b7,
            const float* __restrict__ L1, const float* __restrict__ g8, const float* __restrict__ b8,
            const float* __restrict__ L2, const float* __restrict__ L2b,
            const float* __restrict__ g9, const float* __restrict__ b9,
            const float* __restrict__ L3, const float* __restrict__ L3b,
            float* __restrict__ out) {
  __shared__ float v[1152];
  __shared__ float v2[512];
  __shared__ float v3[256];
  const int b = blockIdx.x, t = threadIdx.x;
  const float inv = rsqrtf(1.0f + 1e-5f);

  for (int i = t; i < 1024; i += 256) v[i] = gmax[(size_t)b * 1024 + i];
  if (t < 64) {
    float a = 0.f;
    #pragma unroll
    for (int c = 0; c < 5; c++) a += l[b * 5 + c] * W6[t * 5 + c];
    a = a * g6[t] * inv + b6[t];
    v[1024 + t] = LRELU(a);
  } else if (t < 128) {
    int o = t - 64;
    float a = 0.f;
    #pragma unroll
    for (int c = 0; c < 7; c++) a += n[b * 7 + c] * W7[o * 7 + c];
    a = a * g7[o] * inv + b7[o];
    v[1088 + o] = LRELU(a);
  }
  __syncthreads();

  for (int o = t; o < 512; o += 256) {
    float a = 0.f;
    for (int c = 0; c < 1152; c++) a += v[c] * L1[(size_t)o * 1152 + c];
    a = a * g8[o] * inv + b8[o];
    v2[o] = fmaxf(a, 0.f);
  }
  __syncthreads();

  {
    float a = 0.f;
    for (int c = 0; c < 512; c++) a += v2[c] * L2[(size_t)t * 512 + c];
    a += L2b[t];
    a = a * g9[t] * inv + b9[t];
    v3[t] = fmaxf(a, 0.f);
  }
  __syncthreads();

  if (t < 28) {
    float a = 0.f;
    for (int c = 0; c < 256; c++) a += v3[c] * L3[t * 256 + c];
    out[b * 28 + t] = a + L3b[t];
  }
}

// ---------------------------------------------------------------------------
extern "C" void kernel_launch(void* const* d_in, const int* in_sizes, int n_in,
                              void* d_out, int out_size, void* d_ws, size_t ws_size,
                              hipStream_t stream) {
  (void)in_sizes; (void)n_in; (void)out_size; (void)ws_size;
  const int B = 4, N = 2048;

  const float* x   = (const float*)d_in[0];
  const float* l   = (const float*)d_in[1];
  const float* nn  = (const float*)d_in[2];
  const float* W1  = (const float*)d_in[3];
  const float* g1  = (const float*)d_in[4];
  const float* b1  = (const float*)d_in[5];
  const float* W2  = (const float*)d_in[6];
  const float* g2  = (const float*)d_in[7];
  const float* b2  = (const float*)d_in[8];
  const float* W3  = (const float*)d_in[9];
  const float* g3  = (const float*)d_in[10];
  const float* b3  = (const float*)d_in[11];
  const float* W4  = (const float*)d_in[12];
  const float* g4  = (const float*)d_in[13];
  const float* b4  = (const float*)d_in[14];
  const float* W5  = (const float*)d_in[15];
  const float* g5  = (const float*)d_in[16];
  const float* b5  = (const float*)d_in[17];
  const float* W6  = (const float*)d_in[18];
  const float* g6  = (const float*)d_in[19];
  const float* b6  = (const float*)d_in[20];
  const float* W7  = (const float*)d_in[21];
  const float* g7  = (const float*)d_in[22];
  const float* b7  = (const float*)d_in[23];
  const float* L1  = (const float*)d_in[24];
  const float* g8  = (const float*)d_in[25];
  const float* b8  = (const float*)d_in[26];
  const float* L2  = (const float*)d_in[27];
  const float* L2b = (const float*)d_in[28];
  const float* g9  = (const float*)d_in[29];
  const float* b9  = (const float*)d_in[30];
  const float* L3  = (const float*)d_in[31];
  const float* L3b = (const float*)d_in[32];

  // workspace layout
  float* ws    = (float*)d_ws;
  float* xt3   = ws;                                  // B*N*3
  float* norms = xt3 + (size_t)B * N * 3;             // B*N
  float* catf  = norms + (size_t)B * N;               // B*N*512 (x1|x2|x3|x4)
  float* gmax  = catf + (size_t)B * N * 512;          // B*1024
  int*   idxb  = (int*)(gmax + (size_t)B * 1024);     // B*N*32 (reused per layer)
  _Float16* wh1 = (_Float16*)(idxb + (size_t)B * N * 32);   // 64*64
  _Float16* wh2 = wh1 + 64 * 64;                            // 64*128
  _Float16* wh3 = wh2 + 64 * 128;                           // 128*128
  _Float16* wh4 = wh3 + 128 * 128;                          // 256*256
  _Float16* wh5 = wh4 + 256 * 256;                          // 1024*512

  const int BN = B * N;

  // one-shot f16 padded weight mirrors (so B-tile staging is a pure DMA)
  wcvt_kernel<<<(64 * 64 + 255) / 256, 256, 0, stream>>>(W1, wh1, 64, 6, 64);
  wcvt_kernel<<<(64 * 128 + 255) / 256, 256, 0, stream>>>(W2, wh2, 64, 128, 128);
  wcvt_kernel<<<(128 * 128 + 255) / 256, 256, 0, stream>>>(W3, wh3, 128, 128, 128);
  wcvt_kernel<<<(256 * 256 + 255) / 256, 256, 0, stream>>>(W4, wh4, 256, 256, 256);
  wcvt_kernel<<<(1024 * 512 + 255) / 256, 256, 0, stream>>>(W5, wh5, 1024, 512, 512);

  // layer 1 (C=3 -> 64)
  prep_kernel<<<(BN + 255) / 256, 256, 0, stream>>>(x, xt3, norms, N);
  knn_kernel<<<dim3(N / 64, B), 64, 0, stream>>>(xt3, 3, 3, norms, idxb, N);
  edgeconv_wmma<6><<<dim3(N / 2, 1, B), 128, 0, stream>>>(
      xt3, 3, idxb, wh1, g1, b1, catf + 0, 512, N);

  // layer 2 (64 -> 64)
  norms_kernel<<<(BN + 255) / 256, 256, 0, stream>>>(catf + 0, 512, 64, norms, N);
  knn_kernel<<<dim3(N / 64, B), 64, 0, stream>>>(catf + 0, 512, 64, norms, idxb, N);
  edgeconv_wmma<128><<<dim3(N / 2, 1, B), 128, 0, stream>>>(
      catf + 0, 512, idxb, wh2, g2, b2, catf + 64, 512, N);

  // layer 3 (64 -> 128)
  norms_kernel<<<(BN + 255) / 256, 256, 0, stream>>>(catf + 64, 512, 64, norms, N);
  knn_kernel<<<dim3(N / 64, B), 64, 0, stream>>>(catf + 64, 512, 64, norms, idxb, N);
  edgeconv_wmma<128><<<dim3(N / 2, 2, B), 128, 0, stream>>>(
      catf + 64, 512, idxb, wh3, g3, b3, catf + 128, 512, N);

  // layer 4 (128 -> 256)
  norms_kernel<<<(BN + 255) / 256, 256, 0, stream>>>(catf + 128, 512, 128, norms, N);
  knn_kernel<<<dim3(N / 64, B), 64, 0, stream>>>(catf + 128, 512, 128, norms, idxb, N);
  edgeconv_wmma<256><<<dim3(N / 2, 4, B), 128, 0, stream>>>(
      catf + 128, 512, idxb, wh4, g4, b4, catf + 256, 512, N);

  // global feature: cat(512) -> 1024, global max over N
  init_neg_kernel<<<(B * 1024 + 255) / 256, 256, 0, stream>>>(gmax, B * 1024);
  gemm5_wmma<<<dim3(BN / 64, 1024 / 64), 128, 0, stream>>>(catf, wh5, g5, b5, gmax, N);

  // head MLP
  head_kernel<<<B, 256, 0, stream>>>(gmax, l, nn, W6, g6, b6, W7, g7, b7,
                                     L1, g8, b8, L2, L2b, g9, b9, L3, L3b,
                                     (float*)d_out);
}
